// OutputBlock_6992206757940
// MI455X (gfx1250) — matrix operand order repaired and checked by reading, not compiled, not run on hardware
//
#include <hip/hip_runtime.h>
#include <hip/hip_bf16.h>

#define EMB  128
#define NRAD 6
#define NT   12

typedef __attribute__((ext_vector_type(16))) __bf16 v16bf;
typedef __attribute__((ext_vector_type(8)))  float  v8f;

union Frag {
  v16bf v;
  uint4 q[2];
};

__device__ __forceinline__ unsigned short f32_to_bf16(float f) {
  unsigned int u = __float_as_uint(f);
  u += 0x7fffu + ((u >> 16) & 1u);   // round-to-nearest-even
  return (unsigned short)(u >> 16);
}

__global__ void zero_kernel(float* __restrict__ p, int n) {
  int i = blockIdx.x * blockDim.x + threadIdx.x;
  if (i < n) p[i] = 0.0f;
}

// Phase 1: tmp = m * (rbf @ Wrbf), scatter-add into t[N,128].
// One wave per edge, 4 channels per lane (float4). Pure bandwidth phase.
__global__ __launch_bounds__(256) void edge_gate_scatter(
    const float* __restrict__ m, const float* __restrict__ rbf,
    const int* __restrict__ dst, const float* __restrict__ Wrbf,
    float* __restrict__ t, int E) {
  __shared__ float Wl[NRAD * EMB];
  for (int i = threadIdx.x; i < NRAD * EMB; i += blockDim.x) Wl[i] = Wrbf[i];
  __syncthreads();

  const int lane = threadIdx.x & 31;
  const int wave = threadIdx.x >> 5;
  const int epb = blockDim.x >> 5;        // edges per block per iter (8)
  const int stride = gridDim.x * epb;

  for (int e = blockIdx.x * epb + wave; e < E; e += stride) {
    if (e + stride < E)
      __builtin_prefetch(&m[(size_t)(e + stride) * EMB + lane * 4], 0, 1);

    const float* rp = rbf + (size_t)e * NRAD;
    const float r0 = rp[0], r1 = rp[1], r2 = rp[2];
    const float r3 = rp[3], r4 = rp[4], r5 = rp[5];
    const int d = dst[e];

    const float4 mv = *reinterpret_cast<const float4*>(&m[(size_t)e * EMB + lane * 4]);
    float* trow = t + (size_t)d * EMB + lane * 4;
    const float mva[4] = {mv.x, mv.y, mv.z, mv.w};
#pragma unroll
    for (int c = 0; c < 4; ++c) {
      const int col = lane * 4 + c;
      const float g = r0 * Wl[0 * EMB + col] + r1 * Wl[1 * EMB + col] +
                      r2 * Wl[2 * EMB + col] + r3 * Wl[3 * EMB + col] +
                      r4 * Wl[4 * EMB + col] + r5 * Wl[5 * EMB + col];
      atomicAdd(trow + c, mva[c] * g);
    }
  }
}

// Phase 2: fused 3-layer swish MLP (bf16 WMMA, f32 accum) + 128->12 projection
// + per-graph atomic readout. One 16-node tile per wave, 4 waves per block.
__global__ __launch_bounds__(128) void node_mlp_wmma(
    const float* __restrict__ t, const int* __restrict__ node_graph,
    const float* __restrict__ W1, const float* __restrict__ b1,
    const float* __restrict__ W2, const float* __restrict__ b2,
    const float* __restrict__ W3, const float* __restrict__ b3,
    const float* __restrict__ Wf, float* __restrict__ out,
    int N, int numTiles) {
  __shared__ unsigned short WT[EMB * EMB];      // transposed weights, bf16 (32 KB)
  __shared__ unsigned short act[4][16 * EMB];   // per-wave activations, bf16 (16 KB)

  const int tid = threadIdx.x;
  const int wave = tid >> 5;
  const int lane = tid & 31;
  const int row = lane & 15;       // A: M-row / B: N-col / D: N-col
  const int half = lane >> 4;

  int tile = blockIdx.x * 4 + wave;
  const bool activeWave = (tile < numTiles);
  if (!activeWave) tile = numTiles - 1;         // dummy but valid work

  // Stage this wave's 16 activation rows as bf16 (A-friendly row-major).
  for (int i = lane; i < 16 * EMB; i += 32) {
    const int r = i >> 7, c = i & (EMB - 1);
    const int node = tile * 16 + r;
    const float v = (node < N) ? t[(size_t)node * EMB + c] : 0.0f;
    act[wave][i] = f32_to_bf16(v);
  }

  const float* Ws[3] = {W1, W2, W3};
  const float* bs[3] = {b1, b2, b3};

  for (int layer = 0; layer < 3; ++layer) {
    __syncthreads();  // all waves done reading previous WT contents
    const float* W = Ws[layer];
    for (int i = tid; i < EMB * EMB; i += 128) {   // coalesced read, transpose to WT[n][k]
      const int k = i >> 7, n = i & (EMB - 1);
      WT[n * EMB + k] = f32_to_bf16(W[i]);
    }
    __syncthreads();

    v8f acc[8];
#pragma unroll
    for (int nt = 0; nt < 8; ++nt) { v8f z = {}; acc[nt] = z; }

#pragma unroll
    for (int kc = 0; kc < 4; ++kc) {
      Frag A;
      const int ka = kc * 32 + half * 8;   // A elem j: K = (j/8)*16 + half*8 + j%8
      A.q[0] = *reinterpret_cast<const uint4*>(&act[wave][row * EMB + ka]);
      A.q[1] = *reinterpret_cast<const uint4*>(&act[wave][row * EMB + ka + 16]);
      const int kb = kc * 32 + half * 8;   // B elem j: K = (j/8)*16 + half*8 + j%8
#pragma unroll
      for (int nt = 0; nt < 8; ++nt) {
        Frag B;
        B.q[0] = *reinterpret_cast<const uint4*>(&WT[(nt * 16 + row) * EMB + kb]);
        B.q[1] = *reinterpret_cast<const uint4*>(&WT[(nt * 16 + row) * EMB + kb + 16]);
        acc[nt] = __builtin_amdgcn_wmma_f32_16x16x32_bf16(
            false, A.v, false, B.v, (short)0, acc[nt], false, false);
      }
    }

    // bias + swish, write back as bf16 (D elem r -> row r+8*half, col row)
    const float* b = bs[layer];
#pragma unroll
    for (int nt = 0; nt < 8; ++nt) {
      const float bb = b[nt * 16 + row];
#pragma unroll
      for (int r = 0; r < 8; ++r) {
        float x = acc[nt][r] + bb;
        x = x / (1.0f + __expf(-x));
        act[wave][(r + half * 8) * EMB + nt * 16 + row] = f32_to_bf16(x);
      }
    }
  }

  // Final projection: Wf [128x12] zero-padded to 16 output cols.
  __syncthreads();
  for (int i = tid; i < EMB * 16; i += 128) {
    const int k = i >> 4, n = i & 15;
    WT[n * EMB + k] = (n < NT) ? f32_to_bf16(Wf[k * NT + n]) : (unsigned short)0;
  }
  __syncthreads();

  v8f P = {};
#pragma unroll
  for (int kc = 0; kc < 4; ++kc) {
    Frag A, B;
    const int ka = kc * 32 + half * 8;
    A.q[0] = *reinterpret_cast<const uint4*>(&act[wave][row * EMB + ka]);
    A.q[1] = *reinterpret_cast<const uint4*>(&act[wave][row * EMB + ka + 16]);
    const int kb = kc * 32 + half * 8;
    B.q[0] = *reinterpret_cast<const uint4*>(&WT[row * EMB + kb]);
    B.q[1] = *reinterpret_cast<const uint4*>(&WT[row * EMB + kb + 16]);
    P = __builtin_amdgcn_wmma_f32_16x16x32_bf16(
        false, A.v, false, B.v, (short)0, P, false, false);
  }

  if (activeWave && row < NT) {
#pragma unroll
    for (int r = 0; r < 8; ++r) {
      const int node = tile * 16 + r + half * 8;
      if (node < N) {
        const int g = node_graph[node];
        atomicAdd(&out[g * NT + row], P[r]);
      }
    }
  }
}

extern "C" void kernel_launch(void* const* d_in, const int* in_sizes, int n_in,
                              void* d_out, int out_size, void* d_ws, size_t ws_size,
                              hipStream_t stream) {
  const float* m          = (const float*)d_in[0];
  const float* rbf        = (const float*)d_in[1];
  const int*   dst        = (const int*)d_in[2];
  const int*   node_graph = (const int*)d_in[3];
  const float* Wrbf       = (const float*)d_in[4];
  const float* W1 = (const float*)d_in[5];
  const float* b1 = (const float*)d_in[6];
  const float* W2 = (const float*)d_in[7];
  const float* b2 = (const float*)d_in[8];
  const float* W3 = (const float*)d_in[9];
  const float* b3 = (const float*)d_in[10];
  const float* Wf = (const float*)d_in[11];

  const int E = in_sizes[2];   // dst length
  const int N = in_sizes[3];   // node_graph length
  float* out = (float*)d_out;
  float* t   = (float*)d_ws;   // N*EMB f32 scatter accumulator

  const int tElems = N * EMB;
  zero_kernel<<<(tElems + 255) / 256, 256, 0, stream>>>(t, tElems);
  zero_kernel<<<(out_size + 255) / 256, 256, 0, stream>>>(out, out_size);

  const int epb = 256 / 32;                   // 8 edges per block
  const int eblocks = (E + epb - 1) / epb;
  edge_gate_scatter<<<eblocks, 256, 0, stream>>>(m, rbf, dst, Wrbf, t, E);

  const int numTiles = (N + 15) / 16;
  const int mblocks = (numTiles + 3) / 4;
  node_mlp_wmma<<<mblocks, 128, 0, stream>>>(t, node_graph, W1, b1, W2, b2,
                                             W3, b3, Wf, out, N, numTiles);
}
